// AxialAttention_13168369729911
// MI455X (gfx1250) — matrix operand order repaired and checked by reading, not compiled, not run on hardware
//
#include <hip/hip_runtime.h>

// ---------------------------------------------------------------------------
// AxialAttention for MI455X (gfx1250): all matmuls via v_wmma_f32_16x16x32_bf16
// B=1, L=384, C=CH=128, H=4, D=32
// Round 2: fragment-packed weights + transposed V so every WMMA operand is
// loaded as two contiguous 16-byte chunks (global_load_b128 / ds_load_b128).
// ---------------------------------------------------------------------------

typedef __attribute__((ext_vector_type(16))) __bf16 v16bf;
typedef __attribute__((ext_vector_type(8)))  float  v8f;

constexpr int   kL     = 384;
constexpr int   kC     = 128;
constexpr int   kCH    = 128;
constexpr int   kH     = 4;
constexpr int   kD     = 32;
constexpr float kScale = 0.17677669529663687f; // 1/sqrt(32)
constexpr float kInf   = 1.0e9f;
constexpr int   kWPack = 4 * 8 * 32 * 16;      // packed elements per 128x128 weight

__device__ inline v8f vzero8() {
  v8f z;
#pragma unroll
  for (int i = 0; i < 8; ++i) z[i] = 0.0f;
  return z;
}

// 16-bit A-matrix 16x32 per-lane element -> K index (ISA 7.12.2):
// lanes 0-15:  e=0..7 -> K=0..7,  e=8..15 -> K=16..23
// lanes 16-31: e=0..7 -> K=8..15, e=8..15 -> K=24..31
// Both 8-element halves are contiguous in K -> 16B contiguous in memory.
__device__ inline int a_klocal(int e, int half) {
  return ((e >> 3) << 4) + half * 8 + (((e >> 1) & 3) << 1) + (e & 1);
}

// Build a v16bf fragment from two contiguous 16-byte chunks.
__device__ inline v16bf load_frag16(const __bf16* p0, const __bf16* p1) {
  v16bf r;
  ((uint4*)&r)[0] = *(const uint4*)p0;
  ((uint4*)&r)[1] = *(const uint4*)p1;
  return r;
}
__device__ inline v16bf load_frag16_lds(const __bf16* p0, const __bf16* p1) {
  v16bf r;
  ((uint4*)&r)[0] = *(const uint4*)p0;
  ((uint4*)&r)[1] = *(const uint4*)p1;
  return r;
}

__device__ inline v8f wmma_bf16(v16bf a, v16bf b, v8f c) {
  return __builtin_amdgcn_wmma_f32_16x16x32_bf16(false, a, false, b,
                                                 (short)0, c, false, false);
}

// ---------------------------------------------------------------------------
// Pack a 128x128 fp32 weight into B-fragment order, bf16:
//   out[(((ks*8 + nt)*32 + lane)*16) + e] = W[(ks*32 + half*8 + (e&7) + (e>>3)*16)*128 + nt*16 + m]
// so each lane's 16 elements are 32 contiguous bytes.
// ---------------------------------------------------------------------------
__global__ __launch_bounds__(256) void pack_w_kernel(
    const float* __restrict__ W, __bf16* __restrict__ out) {
  for (int i = threadIdx.x + blockIdx.x * 256; i < kWPack; i += 256 * gridDim.x) {
    const int e = i & 15, lane = (i >> 4) & 31, nt = (i >> 9) & 7, ks = i >> 12;
    const int half = lane >> 4, m = lane & 15;
    out[i] = (__bf16)W[(ks * 32 + half * 8 + (e & 7) + ((e >> 3) << 4)) * 128 + nt * 16 + m];
  }
}

// ---------------------------------------------------------------------------
// Kernel 1: fused Q/K/V/G projection.
// grid = (24 mtiles, 384 n), block = 128 (wave w owns weight w: q,k,v,g)
// q scaled by 1/sqrt(D); q/k stored [n][h][l][d]; V stored TRANSPOSED
// [n][h][d][l]; gate = sigmoid, bf16 [n][l][ch].
// ---------------------------------------------------------------------------
__global__ __launch_bounds__(128) void proj_kernel(
    const float* __restrict__ x, int transpose,
    const __bf16* __restrict__ Wq, const __bf16* __restrict__ Wk,
    const __bf16* __restrict__ Wv, const __bf16* __restrict__ Wg,
    const float* __restrict__ bg,
    __bf16* __restrict__ qout, __bf16* __restrict__ kout,
    __bf16* __restrict__ vtout, __bf16* __restrict__ gout) {
  __shared__ __bf16 xt[16][136]; // row stride 272B (16B multiple)
  const int mtile = blockIdx.x;
  const int n     = blockIdx.y;
  const int tid   = threadIdx.x;

  // Stage x tile (fp32 -> bf16). transpose: xc[n,l,c] = x[0,l,n,c]
  for (int i = tid; i < 16 * 128; i += 128) {
    const int r = i >> 7, c = i & 127;
    const int l = mtile * 16 + r;
    const float v = transpose ? x[((size_t)l * kL + n) * kC + c]
                              : x[((size_t)n * kL + l) * kC + c];
    xt[r][c] = (__bf16)v;
  }
  __syncthreads();

  const int wave = tid >> 5, lane = tid & 31;
  const int half = lane >> 4, m = lane & 15;
  const __bf16* W = (wave == 0) ? Wq : (wave == 1) ? Wk : (wave == 2) ? Wv : Wg;

  v8f acc[8];
#pragma unroll
  for (int t = 0; t < 8; ++t) acc[t] = vzero8();

  for (int ks = 0; ks < 4; ++ks) {
    const __bf16* ar = &xt[m][ks * 32 + half * 8];
    const v16bf af = load_frag16_lds(ar, ar + 16);
#pragma unroll
    for (int nt = 0; nt < 8; ++nt) {
      const __bf16* wp = W + (((ks * 8 + nt) * 32 + lane) << 4);
      const v16bf bfr = load_frag16(wp, wp + 8);
      acc[nt] = wmma_bf16(af, bfr, acc[nt]);
    }
  }

  // C/D layout: vgpr r -> row M = r + 8*half, lane%16 -> col
  if (wave == 2) {
    // V: transposed store, contiguous along l (16 bytes per tile-column)
#pragma unroll
    for (int nt = 0; nt < 8; ++nt) {
      const int ch = nt * 16 + m;
      const int h = ch >> 5, d = ch & 31;
      alignas(16) __bf16 tmp[8];
#pragma unroll
      for (int r = 0; r < 8; ++r) tmp[r] = (__bf16)acc[nt][r];
      const size_t off =
          (((size_t)n * kH + h) * kD + d) * kL + mtile * 16 + half * 8;
      *(uint4*)(vtout + off) = *(const uint4*)tmp;
    }
  } else {
#pragma unroll
    for (int nt = 0; nt < 8; ++nt) {
      const int ch = nt * 16 + m;
      const int h = ch >> 5, d = ch & 31;
#pragma unroll
      for (int r = 0; r < 8; ++r) {
        const int l = mtile * 16 + r + half * 8;
        const float val = acc[nt][r];
        const size_t hidx = (((size_t)n * kH + h) * kL + l) * kD + d;
        if (wave == 0)      qout[hidx] = (__bf16)(val * kScale);
        else if (wave == 1) kout[hidx] = (__bf16)val;
        else {
          const float s = 1.0f / (1.0f + __expf(-(val + bg[ch])));
          gout[((size_t)n * kL + l) * kCH + ch] = (__bf16)s;
        }
      }
    }
  }
}

// ---------------------------------------------------------------------------
// Kernel 2: attention for one (n, h, 16-row tile). block = 1 wave.
// QK^T (24 WMMA) -> masked scores in LDS -> softmax (unnormalized exp packed
// in place as bf16) -> PV (24 WMMA, V read transposed) -> /= row sums.
// ---------------------------------------------------------------------------
__global__ __launch_bounds__(32) void attn_kernel(
    const __bf16* __restrict__ q, const __bf16* __restrict__ k,
    const __bf16* __restrict__ vt, const float* __restrict__ mask,
    __bf16* __restrict__ o) {
  __shared__ float sraw[16][400]; // row stride 1600B (16B multiple)
  __shared__ float sums[16];
  const int itile = blockIdx.x;
  const int h     = blockIdx.y;
  const int n     = blockIdx.z;
  const int lane = threadIdx.x, half = lane >> 4, m = lane & 15;
  const size_t base = ((size_t)n * kH + h) * kL; // q/k rows for (n,h)

  // Q fragment (A, 16x32), pre-scaled; both halves 16B contiguous
  const __bf16* qr = q + (base + itile * 16 + m) * kD + half * 16;
  const v16bf aq = load_frag16(qr, qr + 8);

  float mi[8];
#pragma unroll
  for (int r = 0; r < 8; ++r) mi[r] = mask[itile * 16 + r + half * 8];

  // scores = q . k^T  (B fragment: lane m = col j, elems contiguous in d)
  for (int jt = 0; jt < 24; ++jt) {
    const __bf16* kr = k + (base + jt * 16 + m) * kD + half * 16;
    const v16bf bk = load_frag16(kr, kr + 8);
    v8f s = wmma_bf16(aq, bk, vzero8());
    const float mj = mask[jt * 16 + m];
#pragma unroll
    for (int r = 0; r < 8; ++r)
      sraw[r + half * 8][jt * 16 + m] = s[r] - kInf * (1.0f - mi[r] * mj);
  }
  __syncthreads();

  // Row softmax: lane<16 owns row `lane`; pack exp(s-max) in place as bf16.
  // (bf16 write offset 2j <= fp32 read offset 4j -> in-place is safe.)
  if (lane < 16) {
    const int row = lane;
    float mx = -3.0e38f;
    for (int j = 0; j < kL; ++j) mx = fmaxf(mx, sraw[row][j]);
    float sum = 0.0f;
    __bf16* prow = (__bf16*)&sraw[row][0];
    for (int j = 0; j < kL; ++j) {
      const float p = __expf(sraw[row][j] - mx);
      sum += p;
      prow[j] = (__bf16)p;
    }
    sums[row] = sum;
  }
  __syncthreads();

  // O = P x V : A = exp-scores [16 x 32j] from LDS; B = vT rows (contig in j)
  const size_t vbase = ((size_t)n * kH + h) * kD;
  v8f o0 = vzero8(), o1 = vzero8();
  for (int jj = 0; jj < 12; ++jj) {
    const __bf16* prow = (const __bf16*)&sraw[m][0] + jj * 32 + half * 16;
    const v16bf ap = load_frag16_lds(prow, prow + 8);
    const __bf16* v0 = vt + (vbase + m) * kL + jj * 32 + half * 16;
    const __bf16* v1 = vt + (vbase + 16 + m) * kL + jj * 32 + half * 16;
    const v16bf bv0 = load_frag16(v0, v0 + 8);
    const v16bf bv1 = load_frag16(v1, v1 + 8);
    o0 = wmma_bf16(ap, bv0, o0);
    o1 = wmma_bf16(ap, bv1, o1);
  }

#pragma unroll
  for (int r = 0; r < 8; ++r) {
    const int row = r + half * 8;
    const float inv = 1.0f / sums[row];
    const int l = itile * 16 + row;
    const size_t ob = ((size_t)n * kL + l) * kCH + h * kD;
    o[ob + m]      = (__bf16)(o0[r] * inv);
    o[ob + 16 + m] = (__bf16)(o1[r] * inv);
  }
}

// ---------------------------------------------------------------------------
// Kernel 3: out = x + (o @ Wo + bo) * gate, with optional transposed scatter.
// grid = (24 mtiles, 384 n), block = 128 (4 waves x 2 N-tiles each)
// ---------------------------------------------------------------------------
__global__ __launch_bounds__(128) void out_kernel(
    const __bf16* __restrict__ o, const __bf16* __restrict__ Wo,
    const float* __restrict__ bo, const __bf16* __restrict__ g,
    const float* __restrict__ xin, float* __restrict__ xout, int transpose) {
  __shared__ __bf16 ot[16][136];
  const int mtile = blockIdx.x;
  const int n     = blockIdx.y;
  const int tid   = threadIdx.x;

  for (int i = tid; i < 16 * 128; i += 128) {
    const int r = i >> 7, c = i & 127;
    ot[r][c] = o[((size_t)n * kL + mtile * 16 + r) * kCH + c];
  }
  __syncthreads();

  const int wave = tid >> 5, lane = tid & 31;
  const int half = lane >> 4, m = lane & 15;

  v8f acc[2] = {vzero8(), vzero8()};
  for (int ks = 0; ks < 4; ++ks) {
    const __bf16* ar = &ot[m][ks * 32 + half * 8];
    const v16bf af = load_frag16_lds(ar, ar + 16);
#pragma unroll
    for (int t = 0; t < 2; ++t) {
      const int nt = wave * 2 + t;
      const __bf16* wp = Wo + (((ks * 8 + nt) * 32 + lane) << 4);
      const v16bf bfr = load_frag16(wp, wp + 8);
      acc[t] = wmma_bf16(af, bfr, acc[t]);
    }
  }

#pragma unroll
  for (int t = 0; t < 2; ++t) {
    const int nt = wave * 2 + t;
    const int c = nt * 16 + m;
    const float bov = bo[c];
#pragma unroll
    for (int r = 0; r < 8; ++r) {
      const int l = mtile * 16 + r + half * 8;
      const float gate = (float)g[((size_t)n * kL + l) * kCH + c];
      const size_t idx = transpose ? ((size_t)l * kL + n) * kC + c
                                   : ((size_t)n * kL + l) * kC + c;
      xout[idx] = xin[idx] + (acc[t][r] + bov) * gate;
    }
  }
}

// ---------------------------------------------------------------------------
// Host launch
// ---------------------------------------------------------------------------
extern "C" void kernel_launch(void* const* d_in, const int* in_sizes, int n_in,
                              void* d_out, int out_size, void* d_ws, size_t ws_size,
                              hipStream_t stream) {
  (void)in_sizes; (void)n_in; (void)out_size;
  // setup_inputs() dict order:
  const float* x      = (const float*)d_in[0];
  const float* mask   = (const float*)d_in[1];
  const float* Wq_row = (const float*)d_in[2];
  const float* Wk_row = (const float*)d_in[3];
  const float* Wv_row = (const float*)d_in[4];
  const float* Wg_row = (const float*)d_in[5];
  const float* Wq_col = (const float*)d_in[6];
  const float* Wk_col = (const float*)d_in[7];
  const float* Wv_col = (const float*)d_in[8];
  const float* Wg_col = (const float*)d_in[9];
  const float* Wo_row = (const float*)d_in[10];
  const float* Wo_col = (const float*)d_in[11];
  const float* bg_row = (const float*)d_in[12];
  const float* bg_col = (const float*)d_in[13];
  const float* bo_row = (const float*)d_in[14];
  const float* bo_col = (const float*)d_in[15];
  float* out = (float*)d_out;

  // Workspace: five bf16 chunks (q, k, vT, gate, o) + 10 packed weights.
  const size_t CHUNK = (size_t)kL * kL * kCH * sizeof(__bf16); // 37.75 MB
  const size_t WPB   = (size_t)kWPack * sizeof(__bf16);        // 32 KB
  if (ws_size < 5 * CHUNK + 10 * WPB) return;
  char* ws = (char*)d_ws;
  __bf16* q  = (__bf16*)(ws);
  __bf16* kk = (__bf16*)(ws + 1 * CHUNK);
  __bf16* vT = (__bf16*)(ws + 2 * CHUNK);
  __bf16* gg = (__bf16*)(ws + 3 * CHUNK);
  __bf16* oo = (__bf16*)(ws + 4 * CHUNK);
  __bf16* wp[10];
  const float* wsrc[10] = {Wq_row, Wk_row, Wv_row, Wg_row, Wo_row,
                           Wq_col, Wk_col, Wv_col, Wg_col, Wo_col};
  for (int i = 0; i < 10; ++i) {
    wp[i] = (__bf16*)(ws + 5 * CHUNK + i * WPB);
    pack_w_kernel<<<16, 256, 0, stream>>>(wsrc[i], wp[i]);
  }

  const dim3 gp(24, 384), bp(128);
  const dim3 ga(24, kH, 384), ba(32);

  // Row attention: x -> out
  proj_kernel<<<gp, bp, 0, stream>>>(x, 0, wp[0], wp[1], wp[2], wp[3],
                                     bg_row, q, kk, vT, gg);
  attn_kernel<<<ga, ba, 0, stream>>>(q, kk, vT, mask, oo);
  out_kernel<<<gp, bp, 0, stream>>>(oo, wp[4], bo_row, gg, x, out, 0);

  // Column attention: out -> out (transposed gather/scatter, elementwise in place)
  proj_kernel<<<gp, bp, 0, stream>>>(out, 1, wp[5], wp[6], wp[7], wp[8],
                                     bg_col, q, kk, vT, gg);
  attn_kernel<<<ga, ba, 0, stream>>>(q, kk, vT, mask, oo);
  out_kernel<<<gp, bp, 0, stream>>>(oo, wp[9], bo_col, gg, out, out, 1);
}
